// MambaModel_40939628265737
// MI455X (gfx1250) — matrix operand order repaired and checked by reading, not compiled, not run on hardware
//
#include <hip/hip_runtime.h>
#include <hip/hip_bf16.h>
#include <math.h>

namespace {

constexpr int kNL    = 4;
constexpr int kDM    = 1024;
constexpr int kDI    = 2048;
constexpr int kDS    = 16;
constexpr int kDTR   = 64;
constexpr int kDConv = 4;
constexpr int kVocab = 364;
constexpr int kB     = 8;
constexpr int kL     = 1024;
constexpr int kM     = kB * kL;        // 8192 rows
constexpr int kXDB   = kDTR + 2 * kDS; // 96
constexpr int kPadE  = 40;             // padded K-stride of LDS tiles (80 B)

typedef __attribute__((ext_vector_type(16))) __bf16 v16bf;
typedef __attribute__((ext_vector_type(8)))  __bf16 v8bf;
typedef __attribute__((ext_vector_type(8)))  float  v8f;

__device__ __forceinline__ float silu_f(float x)     { return x / (1.f + __expf(-x)); }
__device__ __forceinline__ float softplus_f(float x) { return x > 20.f ? x : log1pf(__expf(x)); }

// ---------------------------------------------------------------------------
// Async global->LDS copy of 16 bytes per lane (CDNA5 GLOBAL_LOAD_ASYNC_TO_LDS,
// tracked by ASYNCcnt). Inline asm per CDNA5 ISA.
__device__ __forceinline__ void async_cp16(const __bf16* g, __bf16* l) {
  unsigned loff = (unsigned)(uintptr_t)l;        // generic->LDS: low 32 bits
  unsigned long long gaddr = (unsigned long long)(uintptr_t)g;
  asm volatile("global_load_async_to_lds_b128 %0, %1, off"
               :: "v"(loff), "v"(gaddr) : "memory");
}
__device__ __forceinline__ void wait_async0() {
  asm volatile("s_wait_asynccnt 0x0" ::: "memory");
}

// ---------------------------------------------------------------------------
// WMMA fragment loaders from LDS tiles (bf16, 16x16x32). Layouts per CDNA5
// ISA 7.12.2. A: lane m=lane&15; lanes 0-15 hold K {0-7,16-23}, lanes 16-31
// hold K {8-15,24-31}. B: lane n=lane&15; same K split as A.
__device__ __forceinline__ v16bf ld_frag_a_lds(const __bf16* sA, int row0) {
  const int lane = threadIdx.x & 31;
  const __bf16* p = sA + (row0 + (lane & 15)) * kPadE + ((lane >> 4) << 3);
  v8bf lo = *(const v8bf*)p;
  v8bf hi = *(const v8bf*)(p + 16);
  v16bf r;
#pragma unroll
  for (int i = 0; i < 8; ++i) { r[i] = lo[i]; r[i + 8] = hi[i]; }
  return r;
}
__device__ __forceinline__ v16bf ld_frag_b_lds(const __bf16* sB, int col0) {
  const int lane = threadIdx.x & 31;
  const __bf16* p = sB + (col0 + (lane & 15)) * kPadE + ((lane >> 4) << 3);
  v8bf lo = *(const v8bf*)p;
  v8bf hi = *(const v8bf*)(p + 16);
  v16bf r;
#pragma unroll
  for (int i = 0; i < 8; ++i) { r[i] = lo[i]; r[i + 8] = hi[i]; }
  return r;
}

enum { EPI_NONE = 0, EPI_BIAS = 1, EPI_SPLUS = 2, EPI_RESID = 3 };

// Stage one 32-wide K-slab of the block tile into LDS with async copies.
// A tile: 64x32 bf16 (256 x 16B chunks, 1/thread). B tile: 128x32 (2/thread).
__device__ __forceinline__ void stage_slab(
    const __bf16* __restrict__ A, const __bf16* __restrict__ W,
    int lda, int ldw, int rowBlk, int colBlk, int N, int k0,
    __bf16* sA, __bf16* sB) {
  const int t  = threadIdx.x;
  const int r  = t >> 2;
  const int cc = (t & 3) * 8;            // element offset of 16B chunk
  async_cp16(A + (size_t)(rowBlk + r) * lda + k0 + cc, sA + r * kPadE + cc);
  int c1 = colBlk + r;      if (c1 >= N) c1 = N - 1;   // clamp; stores guarded
  int c2 = colBlk + 64 + r; if (c2 >= N) c2 = N - 1;
  async_cp16(W + (size_t)c1 * ldw + k0 + cc, sB + r * kPadE + cc);
  async_cp16(W + (size_t)c2 * ldw + k0 + cc, sB + (64 + r) * kPadE + cc);
}

// C = A(MxK, bf16) * W^T(KxN from NxK bf16), fp32 accumulate.
// Block tile 64(M) x 128(N), 8 waves, each wave a 32x32 tile (2x2 WMMA accs).
// Double-buffered async global->LDS staging of both operand slabs.
template <int EPI, bool WBF16>
__global__ __launch_bounds__(256) void gemm_bf16_wmma(
    const __bf16* __restrict__ A, const __bf16* __restrict__ W,
    float* __restrict__ Cf, __bf16* __restrict__ Cb,
    const float* __restrict__ bias, const float* __restrict__ resid,
    int M, int N, int K, int lda, int ldw, int ldc) {
  __shared__ __bf16 sA[2][64 * kPadE];
  __shared__ __bf16 sB[2][128 * kPadE];

  const int wave = threadIdx.x >> 5;
  const int lane = threadIdx.x & 31;
  const int rowBlk = blockIdx.y * 64;
  const int colBlk = blockIdx.x * 128;
  const int wrow = (wave >> 2) * 32;     // wave tile within block tile
  const int wcol = (wave & 3) * 32;

  v8f acc[2][2] = {};

  stage_slab(A, W, lda, ldw, rowBlk, colBlk, N, 0, sA[0], sB[0]);
  const int nk = K >> 5;
  for (int i = 0; i < nk; ++i) {
    wait_async0();            // own async issues complete
    __syncthreads();          // everyone's slab visible
    if (i + 1 < nk)
      stage_slab(A, W, lda, ldw, rowBlk, colBlk, N, (i + 1) * 32,
                 sA[(i + 1) & 1], sB[(i + 1) & 1]);
    const __bf16* a = sA[i & 1];
    const __bf16* b = sB[i & 1];
    v16bf a0 = ld_frag_a_lds(a, wrow);
    v16bf a1 = ld_frag_a_lds(a, wrow + 16);
    v16bf b0 = ld_frag_b_lds(b, wcol);
    v16bf b1 = ld_frag_b_lds(b, wcol + 16);
    acc[0][0] = __builtin_amdgcn_wmma_f32_16x16x32_bf16(false, a0, false, b0,
                 (short)0, acc[0][0], false, false);
    acc[0][1] = __builtin_amdgcn_wmma_f32_16x16x32_bf16(false, a0, false, b1,
                 (short)0, acc[0][1], false, false);
    acc[1][0] = __builtin_amdgcn_wmma_f32_16x16x32_bf16(false, a1, false, b0,
                 (short)0, acc[1][0], false, false);
    acc[1][1] = __builtin_amdgcn_wmma_f32_16x16x32_bf16(false, a1, false, b1,
                 (short)0, acc[1][1], false, false);
    __syncthreads();          // all reads done before buffer is re-staged
  }

  // C/D layout: VGPR j -> M = j + 8*(lane>=16), N = lane&15.
#pragma unroll
  for (int ti = 0; ti < 2; ++ti) {
#pragma unroll
    for (int tj = 0; tj < 2; ++tj) {
      const int col   = colBlk + wcol + tj * 16 + (lane & 15);
      const int rbase = rowBlk + wrow + ti * 16 + ((lane >> 4) << 3);
      if (col < N) {
#pragma unroll
        for (int j = 0; j < 8; ++j) {
          float v = acc[ti][tj][j];
          if (EPI == EPI_BIAS)  v += bias[col];
          if (EPI == EPI_SPLUS) v = softplus_f(v + bias[col]);
          if (EPI == EPI_RESID) v += resid[(size_t)(rbase + j) * ldc + col];
          Cf[(size_t)(rbase + j) * ldc + col] = v;
          if (WBF16) Cb[(size_t)(rbase + j) * ldc + col] = (__bf16)v;
        }
      }
    }
  }
}

// ---------------------------------------------------------------------------
__global__ __launch_bounds__(256) void f32_to_bf16(const float* __restrict__ in,
                                                   __bf16* __restrict__ out, int n) {
  int i = blockIdx.x * 256 + threadIdx.x;
  if (i < n) out[i] = (__bf16)in[i];
}

__global__ __launch_bounds__(256) void embed_add(
    const int* __restrict__ x, const int* __restrict__ t,
    const float* __restrict__ emb, const float* __restrict__ temb,
    float* __restrict__ h) {
  int idx  = blockIdx.x * 256 + threadIdx.x;     // < kM*kDM
  int dcol = idx & (kDM - 1);
  int row  = idx >> 10;
  h[idx] = emb[(size_t)x[row] * kDM + dcol] + temb[(size_t)t[row] * kDM + dcol];
}

// One row (D_MODEL=1024) per block; emits bf16 directly for the WMMA GEMM.
__global__ __launch_bounds__(256) void layernorm_bf16(
    const float* __restrict__ h, const float* __restrict__ scale,
    const float* __restrict__ bias, __bf16* __restrict__ out) {
  __shared__ float wsum[8], wsq[8], stats[2];
  const int row = blockIdx.x;
  const float* hp = h + (size_t)row * kDM;
  float s = 0.f, ss = 0.f;
  for (int j = threadIdx.x; j < kDM; j += 256) { float v = hp[j]; s += v; ss += v * v; }
#pragma unroll
  for (int off = 16; off; off >>= 1) {
    s  += __shfl_down(s, off, 32);
    ss += __shfl_down(ss, off, 32);
  }
  if ((threadIdx.x & 31) == 0) { wsum[threadIdx.x >> 5] = s; wsq[threadIdx.x >> 5] = ss; }
  __syncthreads();
  if (threadIdx.x == 0) {
    float t = 0.f, tt = 0.f;
#pragma unroll
    for (int i = 0; i < 8; ++i) { t += wsum[i]; tt += wsq[i]; }
    float mu  = t / kDM;
    float var = tt / kDM - mu * mu;
    stats[0] = mu; stats[1] = rsqrtf(var + 1e-5f);
  }
  __syncthreads();
  const float mu = stats[0], rstd = stats[1];
  for (int j = threadIdx.x; j < kDM; j += 256)
    out[(size_t)row * kDM + j] = (__bf16)((hp[j] - mu) * rstd * scale[j] + bias[j]);
}

// Depthwise causal conv (D_CONV=4) + bias + SiLU. Reads xc = xz[:, :, 0:2048].
__global__ __launch_bounds__(256) void conv_silu(
    const float* __restrict__ xz, const float* __restrict__ cw,
    const float* __restrict__ cb, float* __restrict__ xcf,
    __bf16* __restrict__ xcb) {
  int idx = blockIdx.x * 256 + threadIdx.x;      // < kM*kDI (2^24)
  int c = idx & (kDI - 1);
  int l = (idx >> 11) & (kL - 1);
  int b = idx >> 21;
  float acc = cb[c];
#pragma unroll
  for (int k = 0; k < kDConv; ++k) {
    int ll = l + k - (kDConv - 1);
    if (ll >= 0)
      acc += xz[((size_t)((b << 10) + ll)) * (2 * kDI) + c] * cw[c * kDConv + k];
  }
  float v = silu_f(acc);
  xcf[idx] = v;
  xcb[idx] = (__bf16)v;
}

// Selective scan: one thread per (b,d) chain; 16 states in registers;
// B/C tiles staged in LDS per 16 timesteps; fuses D-skip + SiLU(z) gate.
__global__ __launch_bounds__(256) void selective_scan(
    const float* __restrict__ dt, const float* __restrict__ xdb,
    const float* __restrict__ xconv, const float* __restrict__ xz,
    const float* __restrict__ Alog, const float* __restrict__ Dsk,
    __bf16* __restrict__ yg) {
  const int b = blockIdx.y;
  const int d = blockIdx.x * 256 + threadIdx.x;
  __shared__ float Bs[16][kDS];
  __shared__ float Cs[16][kDS];
  float Aj[kDS], s[kDS];
#pragma unroll
  for (int j = 0; j < kDS; ++j) { Aj[j] = -__expf(Alog[(size_t)d * kDS + j]); s[j] = 0.f; }
  const float dsk = Dsk[d];

  for (int lb = 0; lb < kL; lb += 16) {
    __syncthreads();
    for (int t = threadIdx.x; t < 16 * 2 * kDS; t += 256) {
      int l = t >> 5, j = t & 31;
      size_t row = (size_t)((b << 10) + lb + l);
      if (j < kDS) Bs[l][j]       = xdb[row * kXDB + kDTR + j];
      else         Cs[l][j - kDS] = xdb[row * kXDB + kDTR + kDS + (j - kDS)];
    }
    __syncthreads();
#pragma unroll 4
    for (int l = 0; l < 16; ++l) {
      size_t row = (size_t)((b << 10) + lb + l);
      float dtv = dt[row * kDI + d];
      float u   = xconv[row * kDI + d];
      float zv  = xz[row * (2 * kDI) + kDI + d];
      float du  = dtv * u;
      float y   = 0.f;
#pragma unroll
      for (int j = 0; j < kDS; ++j) {
        float dA = __expf(dtv * Aj[j]);
        s[j] = s[j] * dA + du * Bs[l][j];
        y += s[j] * Cs[l][j];
      }
      float g = (y + dsk * u) * silu_f(zv);
      yg[row * kDI + d] = (__bf16)g;
    }
  }
}

}  // namespace

// ---------------------------------------------------------------------------
extern "C" void kernel_launch(void* const* d_in, const int* in_sizes, int n_in,
                              void* d_out, int out_size, void* d_ws, size_t ws_size,
                              hipStream_t stream) {
  (void)in_sizes; (void)n_in; (void)out_size; (void)ws_size;
  const int*   x        = (const int*)d_in[0];
  const int*   turn     = (const int*)d_in[1];
  const float* emb      = (const float*)d_in[2];
  const float* temb     = (const float*)d_in[3];
  const float* in_w     = (const float*)d_in[4];
  const float* conv_w   = (const float*)d_in[5];
  const float* conv_b   = (const float*)d_in[6];
  const float* x_w      = (const float*)d_in[7];
  const float* dt_w     = (const float*)d_in[8];
  const float* dt_b     = (const float*)d_in[9];
  const float* A_log    = (const float*)d_in[10];
  const float* D_skip   = (const float*)d_in[11];
  const float* out_w    = (const float*)d_in[12];
  const float* ln_s     = (const float*)d_in[13];
  const float* ln_b     = (const float*)d_in[14];
  const float* head_w   = (const float*)d_in[15];
  const float* head_b   = (const float*)d_in[16];
  float* logits = (float*)d_out;

  char* ws = (char*)d_ws;
  size_t off = 0;
  auto alloc = [&](size_t bytes) -> void* {
    void* p = ws + off;
    off = (off + bytes + 255) & ~(size_t)255;
    return p;
  };

  float*  h       = (float*)alloc((size_t)kM * kDM * 4);
  __bf16* hn_b    = (__bf16*)alloc((size_t)kM * kDM * 2);
  float*  xzbuf   = (float*)alloc((size_t)kM * 2 * kDI * 4);
  float*  xconv_f = (float*)alloc((size_t)kM * kDI * 4);
  __bf16* xconv_b = (__bf16*)alloc((size_t)kM * kDI * 2);
  float*  xdb_f   = (float*)alloc((size_t)kM * kXDB * 4);
  __bf16* xdb_b   = (__bf16*)alloc((size_t)kM * kXDB * 2);
  float*  dtbuf   = (float*)alloc((size_t)kM * kDI * 4);
  __bf16* yg_b    = (__bf16*)alloc((size_t)kM * kDI * 2);
  __bf16* h_b     = (__bf16*)alloc((size_t)kM * kDM * 2);
  __bf16* w_in_b  = (__bf16*)alloc((size_t)kNL * 2 * kDI * kDM * 2);
  __bf16* w_x_b   = (__bf16*)alloc((size_t)kNL * kXDB * kDI * 2);
  __bf16* w_dt_b  = (__bf16*)alloc((size_t)kNL * kDI * kDTR * 2);
  __bf16* w_out_b = (__bf16*)alloc((size_t)kNL * kDM * kDI * 2);
  __bf16* w_hd_b  = (__bf16*)alloc((size_t)kVocab * kDM * 2);

  auto cvt = [&](const float* src, __bf16* dst, int n) {
    f32_to_bf16<<<(n + 255) / 256, 256, 0, stream>>>(src, dst, n);
  };
  // Weights -> bf16 (~55 MB total; resident in the 192 MB L2).
  cvt(in_w,  w_in_b,  kNL * 2 * kDI * kDM);
  cvt(x_w,   w_x_b,   kNL * kXDB * kDI);
  cvt(dt_w,  w_dt_b,  kNL * kDI * kDTR);
  cvt(out_w, w_out_b, kNL * kDM * kDI);
  cvt(head_w, w_hd_b, kVocab * kDM);

  embed_add<<<(kM * kDM) / 256, 256, 0, stream>>>(x, turn, emb, temb, h);

  const dim3 blk(256);
  for (int i = 0; i < kNL; ++i) {
    layernorm_bf16<<<kM, blk, 0, stream>>>(h, ln_s + i * kDM, ln_b + i * kDM, hn_b);

    // in_proj: (8192x1024) x (4096x1024)^T -> xz (8192x4096)
    gemm_bf16_wmma<EPI_NONE, false><<<dim3((2 * kDI + 127) / 128, kM / 64), blk, 0, stream>>>(
        hn_b, w_in_b + (size_t)i * 2 * kDI * kDM, xzbuf, nullptr, nullptr, nullptr,
        kM, 2 * kDI, kDM, kDM, kDM, 2 * kDI);

    conv_silu<<<(kM * kDI) / 256, blk, 0, stream>>>(
        xzbuf, conv_w + i * kDI * kDConv, conv_b + i * kDI, xconv_f, xconv_b);

    // x_proj: (8192x2048) x (96x2048)^T -> xdb (8192x96), f32 + bf16
    gemm_bf16_wmma<EPI_NONE, true><<<dim3((kXDB + 127) / 128, kM / 64), blk, 0, stream>>>(
        xconv_b, w_x_b + (size_t)i * kXDB * kDI, xdb_f, xdb_b, nullptr, nullptr,
        kM, kXDB, kDI, kDI, kDI, kXDB);

    // dt_proj + softplus: (8192x64) x (2048x64)^T -> dt (8192x2048)
    gemm_bf16_wmma<EPI_SPLUS, false><<<dim3((kDI + 127) / 128, kM / 64), blk, 0, stream>>>(
        xdb_b, w_dt_b + (size_t)i * kDI * kDTR, dtbuf, nullptr, dt_b + i * kDI, nullptr,
        kM, kDI, kDTR, kXDB, kDTR, kDI);

    selective_scan<<<dim3(kDI / 256, kB), blk, 0, stream>>>(
        dtbuf, xdb_f, xconv_f, xzbuf, A_log + (size_t)i * kDI * kDS,
        D_skip + i * kDI, yg_b);

    // out_proj + residual: (8192x2048) x (1024x2048)^T + h -> h
    gemm_bf16_wmma<EPI_RESID, false><<<dim3((kDM + 127) / 128, kM / 64), blk, 0, stream>>>(
        yg_b, w_out_b + (size_t)i * kDM * kDI, h, nullptr, nullptr, h,
        kM, kDM, kDI, kDI, kDI, kDM);
  }

  cvt(h, h_b, kM * kDM);
  // head: (8192x1024) x (364x1024)^T + bias -> logits (8192x364)
  gemm_bf16_wmma<EPI_BIAS, false><<<dim3((kVocab + 127) / 128, kM / 64), blk, 0, stream>>>(
      h_b, w_hd_b, logits, nullptr, head_b, nullptr,
      kM, kVocab, kDM, kDM, kDM, kVocab);
}
